// Nica_75127567941769
// MI455X (gfx1250) — compile-verified, matrix operand order checked by reading, not run on hardware
//
#include <hip/hip_runtime.h>
#include <hip/hip_bf16.h>

// ---------------- CDNA5 WMMA / TDM types ----------------
typedef __attribute__((ext_vector_type(16))) __bf16 v16bf;
typedef __attribute__((ext_vector_type(8)))  float  v8f;
typedef __attribute__((ext_vector_type(8)))  float  v8ff;
typedef __attribute__((ext_vector_type(8)))  __bf16 v8bf;
typedef __attribute__((ext_vector_type(4)))  unsigned int u32x4;
typedef __attribute__((ext_vector_type(8)))  int i32x8;
typedef __attribute__((ext_vector_type(4)))  int i32x4;

#ifndef __has_builtin
#define __has_builtin(x) 0
#endif
#if defined(__gfx1250__) && __has_builtin(__builtin_amdgcn_tensor_load_to_lds) && \
    __has_builtin(__builtin_amdgcn_s_wait_tensorcnt)
#define USE_TDM 1
#else
#define USE_TDM 0
#endif

// Problem constants
#define BB 128
#define PP 196
#define ENCD 2048
#define AA 512
#define EE 512
#define HH 512
#define VV 10000
#define MAXLEN 52
#define TT 51           // MAXLEN-1 decode steps

// Output flat offsets (floats): preds | caps | alphas | sort_ind
#define PRED_OFF   0L
#define CAPS_OFF   (PRED_OFF + (long)BB*TT*VV)
#define ALPHA_OFF  (CAPS_OFF + (long)BB*MAXLEN)
#define SIND_OFF   (ALPHA_OFF + (long)BB*TT*PP)

__device__ __forceinline__ float sigf(float x){ return 1.0f/(1.0f+__expf(-x)); }

#if USE_TDM
// Issue one TDM load of a 64-row x 128-elem bf16 tile of W (row pitch K elems)
// into LDS at dst, padded to 272B rows (pad_interval=256B, pad_amount=4 DWORDs).
__device__ __forceinline__ void tdm_load_tile(const __bf16* W, int K, int n0, int ch, __bf16* dst)
{
    unsigned long long ga = (unsigned long long)(uintptr_t)W +
        ((unsigned long long)(unsigned)n0 * (unsigned)K + ((unsigned)ch << 7)) * 2ull;
    u32x4 g0;
    g0[0] = 1u;                                   // count=1 (valid user descriptor)
    g0[1] = (unsigned)(uintptr_t)dst;             // lds_addr (low 32 bits of flat = LDS offset)
    g0[2] = (unsigned)ga;                         // global_addr[31:0]
    g0[3] = (unsigned)(ga >> 32) | 0x80000000u;   // global_addr[56:32] | type=2 ("image")
    i32x8 g1;
    g1[0] = (int)((1u<<16) | (1u<<20) | (5u<<22) | (3u<<25)); // data_size=2B, pad_en, every 256B, +4 DWORDs
    g1[1] = (int)(128u << 16);                    // tensor_dim0 = 128 (bits 63:48 = low16)
    g1[2] = (int)(64u  << 16);                    // tensor_dim0 hi=0 | tensor_dim1 = 64
    g1[3] = (int)(128u << 16);                    // tensor_dim1 hi=0 | tile_dim0 = 128
    g1[4] = 64;                                   // tile_dim1 = 64, tile_dim2 = 0
    g1[5] = K;                                    // tensor_dim0_stride low32 (elements)
    g1[6] = 0;
    g1[7] = 0;
    i32x4 gz4 = {0, 0, 0, 0};
    i32x8 gz8 = {0, 0, 0, 0, 0, 0, 0, 0};
    __builtin_amdgcn_tensor_load_to_lds(g0, g1, gz4, gz4, gz8, 0);
}
#endif

// ================== generic bf16-WMMA GEMM ==================
// C[m,n] = sum_k A[m,k]*W[n,k] + bias[n] (+ addC[m*ldadd+n]) ; optional mask zeroing.
// A: fp32 (converted on the fly), W: bf16 [N,K] row-major. K%32==0, N%16==0.
// Block: 256 threads = 8 waves -> 128 rows x 64 cols per block.
__global__ void __launch_bounds__(256)
k_gemm_bf16(const float* __restrict__ A, long lda,
            const int* __restrict__ rowmap, int rpg,
            const __bf16* __restrict__ W, const float* __restrict__ bias,
            float* __restrict__ C, long ldc,
            const float* __restrict__ addC, long ldadd,
            const int* __restrict__ mask_dl, int step,
            int M, int N, int K)
{
#if USE_TDM
    __shared__ __align__(16) __bf16 lds_w[2][64 * 136];   // 272B row pitch, double buffered
#endif
    const int lane  = threadIdx.x & 31;
    const int wave  = threadIdx.x >> 5;
    const int m0    = blockIdx.y * 128 + wave * 16;
    const int n0    = blockIdx.x * 64;

    const int lrow  = lane & 15;              // row-in-tile for A, col-in-tile for B/C
    const int khalf = (lane >> 4) * 8;        // K sub-offset per half-wave

    const bool full_n = (n0 + 64 <= N);
#if USE_TDM
    const bool use_tdm = full_n && ((K & 127) == 0) && ((M & 127) == 0);
    if (!use_tdm && m0 >= M) return;
#else
    if (m0 >= M) return;
#endif

    long arow_idx = (m0 < M ? m0 : 0) + lrow;
    if (rowmap) {
        long grp    = arow_idx / rpg;
        long within = arow_idx - grp * rpg;
        arow_idx    = (long)rowmap[grp] * rpg + within;
    }
    const float* arow = A + arow_idx * lda;

    const __bf16* wrow[4];
#pragma unroll
    for (int j = 0; j < 4; j++)
        wrow[j] = W + (long)(n0 + j * 16 + lrow) * K;

    v8f acc[4] = {};

#if USE_TDM
    if (use_tdm) {
        // -------- TDM double-buffered path: weights staged to LDS via Tensor Data Mover --------
        if (wave == 0) {
            tdm_load_tile(W, K, n0, 0, &lds_w[0][0]);
            __builtin_amdgcn_s_wait_tensorcnt((short)0);
        }
        __syncthreads();
        const int nch = K >> 7;                       // 128-element K chunks
        for (int ch = 0; ch < nch; ch++) {
            const int cur = ch & 1;
            if (wave == 0 && ch + 1 < nch)
                tdm_load_tile(W, K, n0, ch + 1, &lds_w[cur ^ 1][0]);
            const __bf16* lw = &lds_w[cur][0];
#pragma unroll
            for (int ks = 0; ks < 4; ks++) {
                const int k0 = (ch << 7) + (ks << 5);
                v8ff a_lo = *(const v8ff*)(arow + k0 + khalf);
                v8ff a_hi = *(const v8ff*)(arow + k0 + khalf + 16);
                v16bf af;
#pragma unroll
                for (int i = 0; i < 8; i++) { af[i] = (__bf16)a_lo[i]; af[8+i] = (__bf16)a_hi[i]; }
                __builtin_prefetch(arow + k0 + 64, 0, 1);
#pragma unroll
                for (int j = 0; j < 4; j++) {
                    const __bf16* wr = lw + (j * 16 + lrow) * 136 + (ks << 5) + khalf;
                    v8bf b_lo = *(const v8bf*)(wr);
                    v8bf b_hi = *(const v8bf*)(wr + 16);
                    v16bf bf = __builtin_shufflevector(b_lo, b_hi,
                        0,1,2,3,4,5,6,7,8,9,10,11,12,13,14,15);
                    acc[j] = __builtin_amdgcn_wmma_f32_16x16x32_bf16(
                                 false, af, false, bf, (short)0, acc[j], false, false);
                }
            }
            if (wave == 0 && ch + 1 < nch) __builtin_amdgcn_s_wait_tensorcnt((short)0);
            __syncthreads();
        }
    } else
#endif
    if (full_n) {
        // -------- direct-global path, all 4 N-tiles valid: branch-free inner loop --------
        for (int k0 = 0; k0 < K; k0 += 32) {
            v8ff a_lo = *(const v8ff*)(arow + k0 + khalf);
            v8ff a_hi = *(const v8ff*)(arow + k0 + khalf + 16);
            v16bf af;
#pragma unroll
            for (int i = 0; i < 8; i++) { af[i] = (__bf16)a_lo[i]; af[8+i] = (__bf16)a_hi[i]; }
            __builtin_prefetch(arow + k0 + 64, 0, 1);
#pragma unroll
            for (int j = 0; j < 4; j++) {
                v8bf b_lo = *(const v8bf*)(wrow[j] + k0 + khalf);
                v8bf b_hi = *(const v8bf*)(wrow[j] + k0 + khalf + 16);
                v16bf bf = __builtin_shufflevector(b_lo, b_hi,
                    0,1,2,3,4,5,6,7,8,9,10,11,12,13,14,15);
                acc[j] = __builtin_amdgcn_wmma_f32_16x16x32_bf16(
                             false, af, false, bf, (short)0, acc[j], false, false);
            }
        }
    } else {
        // -------- N-edge path (only the last preds block, N%64 != 0) --------
        for (int k0 = 0; k0 < K; k0 += 32) {
            v8ff a_lo = *(const v8ff*)(arow + k0 + khalf);
            v8ff a_hi = *(const v8ff*)(arow + k0 + khalf + 16);
            v16bf af;
#pragma unroll
            for (int i = 0; i < 8; i++) { af[i] = (__bf16)a_lo[i]; af[8+i] = (__bf16)a_hi[i]; }
#pragma unroll
            for (int j = 0; j < 4; j++) {
                if (n0 + j * 16 < N) {  // block-uniform
                    v8bf b_lo = *(const v8bf*)(wrow[j] + k0 + khalf);
                    v8bf b_hi = *(const v8bf*)(wrow[j] + k0 + khalf + 16);
                    v16bf bf = __builtin_shufflevector(b_lo, b_hi,
                        0,1,2,3,4,5,6,7,8,9,10,11,12,13,14,15);
                    acc[j] = __builtin_amdgcn_wmma_f32_16x16x32_bf16(
                                 false, af, false, bf, (short)0, acc[j], false, false);
                }
            }
        }
    }

    // C fragment: col n = lane&15, row m = r + (lane>=16)*8
    if (m0 >= M) return;
    const int lhalf = lane >> 4;
#pragma unroll
    for (int j = 0; j < 4; j++) {
        if (n0 + j * 16 >= N) continue;
        int n = n0 + j * 16 + lrow;
        float bb = bias ? bias[n] : 0.0f;
#pragma unroll
        for (int r = 0; r < 8; r++) {
            int m = m0 + lhalf * 8 + r;
            if (m < M) {
                float v = acc[j][r] + bb;
                if (addC) v += addC[(long)m * ldadd + n];
                if (mask_dl && step >= mask_dl[m]) v = 0.0f;
                C[(long)m * ldc + n] = v;
            }
        }
    }
}

// ================== setup kernels ==================
__global__ void k_sort(const int* __restrict__ lens, const int* __restrict__ caps,
                       int* sort_ind, int* dl, int* caps_sorted,
                       float* out_caps, float* out_sind)
{
    int i  = threadIdx.x;                 // 0..127
    int li = lens[i];
    int rank = 0;
    for (int j = 0; j < BB; j++) {
        int lj = lens[j];
        if (lj > li || (lj == li && j < i)) rank++;
    }
    sort_ind[rank] = i;
    dl[rank]       = li - 1;
    out_sind[rank] = (float)i;
    __threadfence_block();
    __syncthreads();
    int b  = i;
    int si = sort_ind[b];
    for (int t = 0; t < MAXLEN; t++) {
        int cv = caps[si * MAXLEN + t];
        caps_sorted[b * MAXLEN + t] = cv;
        out_caps[b * MAXLEN + t]    = (float)cv;
    }
}

__global__ void k_f32_to_bf16(const float* __restrict__ s, __bf16* __restrict__ d, long n){
    long i = (long)blockIdx.x * blockDim.x + threadIdx.x;
    if (i < n) d[i] = (__bf16)s[i];
}

__global__ void k_concat2_bf16(const float* __restrict__ a, int ka,
                               const float* __restrict__ b, int kb,
                               __bf16* __restrict__ d, long total){
    long i = (long)blockIdx.x * blockDim.x + threadIdx.x;
    if (i >= total) return;
    int K = ka + kb;
    long r = i / K; int c = (int)(i - r * K);
    float v = (c < ka) ? a[r * ka + c] : b[r * kb + (c - ka)];
    d[i] = (__bf16)v;
}

__global__ void k_addvec(const float* a, const float* b, float* d, int n){
    int i = blockIdx.x * blockDim.x + threadIdx.x;
    if (i < n) d[i] = a[i] + b[i];
}

__global__ void k_embgather(const float* __restrict__ emb, const int* __restrict__ caps_sorted,
                            float* __restrict__ emb_seq){
    long idx = (long)blockIdx.x * blockDim.x + threadIdx.x;
    if (idx >= (long)BB * TT * EE) return;
    int e = (int)(idx % EE);
    long r = idx / EE;
    int t = (int)(r % TT);
    int b = (int)(r / TT);
    int tok = caps_sorted[b * MAXLEN + t];
    emb_seq[idx] = emb[(long)tok * EE + e];
}

__global__ void k_mean(const float* __restrict__ enc, const int* __restrict__ sind,
                       float* __restrict__ mean){
    int b = blockIdx.x;
    const float* base = enc + (long)sind[b] * PP * ENCD;
    for (int k = threadIdx.x; k < ENCD; k += blockDim.x) {
        float s = 0.f;
        for (int p = 0; p < PP; p++) s += base[(long)p * ENCD + k];
        mean[(long)b * ENCD + k] = s * (1.0f / (float)PP);
    }
}

// ================== per-step kernels ==================
__global__ void k_score(const float* __restrict__ att1, const float* __restrict__ att2,
                        const float* __restrict__ faw, const float* __restrict__ fab,
                        float* __restrict__ score){
    long gid = (long)blockIdx.x * blockDim.x + threadIdx.x;
    int wid  = (int)(gid >> 5);
    int lane = threadIdx.x & 31;
    if (wid >= BB * PP) return;
    int b = wid / PP;
    const float* r1 = att1 + (long)wid * AA;
    const float* r2 = att2 + (long)b * AA;
    float s = 0.f;
    for (int a = lane; a < AA; a += 32) {
        float v = r1[a] + r2[a];
        v = v > 0.f ? v : 0.f;
        s += v * faw[a];
    }
    for (int off = 16; off; off >>= 1) s += __shfl_xor(s, off, 32);
    if (lane == 0) score[wid] = s + fab[0];
}

__global__ void k_softmax(const float* __restrict__ score, float* __restrict__ alpha,
                          float* __restrict__ out_alpha, const int* __restrict__ dl, int t){
    __shared__ float red[256];
    int b = blockIdx.x, tid = threadIdx.x;
    float v = (tid < PP) ? score[b * PP + tid] : -1e30f;
    red[tid] = v; __syncthreads();
    for (int s = 128; s; s >>= 1) { if (tid < s) red[tid] = fmaxf(red[tid], red[tid + s]); __syncthreads(); }
    float mx = red[0]; __syncthreads();
    float e = (tid < PP) ? __expf(v - mx) : 0.f;
    red[tid] = e; __syncthreads();
    for (int s = 128; s; s >>= 1) { if (tid < s) red[tid] += red[tid + s]; __syncthreads(); }
    float inv = 1.0f / red[0];
    if (tid < PP) {
        float a = e * inv;
        alpha[b * PP + tid] = a;
        bool m = t < dl[b];
        out_alpha[((long)b * TT + t) * PP + tid] = m ? a : 0.f;
    }
}

__global__ void k_zctx(const float* __restrict__ enc, const int* __restrict__ sind,
                       const float* __restrict__ alpha, float* __restrict__ zraw){
    __shared__ float sal[PP];
    int b = blockIdx.x, tid = threadIdx.x;
    if (tid < PP) sal[tid] = alpha[b * PP + tid];
    __syncthreads();
    int k = blockIdx.y * 256 + tid;
    const float* base = enc + (long)sind[b] * PP * ENCD + k;
    float s = 0.f;
    for (int p = 0; p < PP; p++) s += base[(long)p * ENCD] * sal[p];
    zraw[(long)b * ENCD + k] = s;
}

__global__ void k_fuse1(const float* __restrict__ emb_seq, int t,
                        const float* __restrict__ betapre, const float* __restrict__ zraw,
                        const float* __restrict__ h,
                        float* __restrict__ xcat, float* __restrict__ precat){
    long idx = (long)blockIdx.x * blockDim.x + threadIdx.x;
    if (idx >= (long)BB * (EE + ENCD + HH)) return;
    int col = (int)(idx % (EE + ENCD + HH));
    int b   = (int)(idx / (EE + ENCD + HH));
    float v;
    if (col < EE) {
        v = emb_seq[((long)b * TT + t) * EE + col];
    } else if (col < EE + ENCD) {
        int k = col - EE;
        v = sigf(betapre[(long)b * ENCD + k]) * zraw[(long)b * ENCD + k];
        precat[(long)b * (HH + ENCD) + col] = v;   // precat = [h_new(512) | z(2048)]
    } else {
        v = h[(long)b * HH + (col - EE - ENCD)];
    }
    xcat[idx] = v;
}

__global__ void k_cell(const float* __restrict__ gates, float* __restrict__ h,
                       float* __restrict__ c, float* __restrict__ precat,
                       const int* __restrict__ dl, int t){
    long idx = (long)blockIdx.x * blockDim.x + threadIdx.x;
    if (idx >= (long)BB * HH) return;
    int j = (int)(idx % HH);
    int b = (int)(idx / HH);
    const float* g = gates + (long)b * 4 * HH;
    float si = sigf(g[j]);
    float sf = sigf(g[HH + j]);
    float gg = tanhf(g[2 * HH + j]);
    float so = sigf(g[3 * HH + j]);
    float cn = sf * c[idx] + si * gg;
    float hn = so * tanhf(cn);
    precat[(long)b * (HH + ENCD) + j] = hn;
    if (t < dl[b]) { h[idx] = hn; c[idx] = cn; }
}

// ================== host side ==================
static inline size_t aln256(size_t x){ return (x + 255) & ~(size_t)255; }

extern "C" void kernel_launch(void* const* d_in, const int* in_sizes, int n_in,
                              void* d_out, int out_size, void* d_ws, size_t ws_size,
                              hipStream_t stream)
{
    (void)in_sizes; (void)n_in; (void)out_size; (void)ws_size;
    const float* enc      = (const float*)d_in[0];
    const int*   caps     = (const int*)  d_in[1];
    const int*   lens     = (const int*)  d_in[2];
    const float* enc_att_w= (const float*)d_in[3];  const float* enc_att_b= (const float*)d_in[4];
    const float* dec_att_w= (const float*)d_in[5];  const float* dec_att_b= (const float*)d_in[6];
    const float* faw      = (const float*)d_in[7];  const float* fab      = (const float*)d_in[8];
    const float* emb      = (const float*)d_in[9];
    const float* w_ih     = (const float*)d_in[10]; const float* b_ih     = (const float*)d_in[11];
    const float* w_hh     = (const float*)d_in[12]; const float* b_hh     = (const float*)d_in[13];
    const float* init_h_w = (const float*)d_in[14]; const float* init_h_b = (const float*)d_in[15];
    const float* init_c_w = (const float*)d_in[16]; const float* init_c_b = (const float*)d_in[17];
    const float* lh_w     = (const float*)d_in[18]; const float* lh_b     = (const float*)d_in[19];
    const float* lz_w     = (const float*)d_in[20]; const float* lz_b     = (const float*)d_in[21];
    const float* fbeta_w  = (const float*)d_in[22]; const float* fbeta_b  = (const float*)d_in[23];
    const float* fc_w     = (const float*)d_in[24]; const float* fc_b     = (const float*)d_in[25];

    float* out = (float*)d_out;

    char* base = (char*)d_ws; size_t off = 0;
    auto alloc = [&](size_t bytes)->void* { void* r = base + off; off = aln256(off + bytes); return r; };

    __bf16* wb_enc_att = (__bf16*)alloc((size_t)AA * ENCD * 2);
    __bf16* wb_dec_att = (__bf16*)alloc((size_t)AA * HH * 2);
    __bf16* wb_init_h  = (__bf16*)alloc((size_t)HH * ENCD * 2);
    __bf16* wb_init_c  = (__bf16*)alloc((size_t)HH * ENCD * 2);
    __bf16* wb_fbeta   = (__bf16*)alloc((size_t)ENCD * HH * 2);
    __bf16* wb_lstm    = (__bf16*)alloc((size_t)(4*HH) * (EE + ENCD + HH) * 2);
    __bf16* wb_out     = (__bf16*)alloc((size_t)EE * (HH + ENCD) * 2);
    __bf16* wb_fc      = (__bf16*)alloc((size_t)VV * HH * 2);

    float* b_lstm   = (float*)alloc((size_t)4*HH*4);
    float* b_out    = (float*)alloc((size_t)EE*4);
    float* mean_enc = (float*)alloc((size_t)BB*ENCD*4);
    float* emb_seq  = (float*)alloc((size_t)BB*TT*EE*4);
    float* att1     = (float*)alloc((size_t)BB*PP*AA*4);
    float* hbuf     = (float*)alloc((size_t)BB*HH*4);
    float* cbuf     = (float*)alloc((size_t)BB*HH*4);
    float* att2     = (float*)alloc((size_t)BB*AA*4);
    float* score    = (float*)alloc((size_t)BB*PP*4);
    float* alpha    = (float*)alloc((size_t)BB*PP*4);
    float* zraw     = (float*)alloc((size_t)BB*ENCD*4);
    float* betapre  = (float*)alloc((size_t)BB*ENCD*4);
    float* xcat     = (float*)alloc((size_t)BB*(EE+ENCD+HH)*4);
    float* precat   = (float*)alloc((size_t)BB*(HH+ENCD)*4);
    float* gates    = (float*)alloc((size_t)BB*4*HH*4);
    float* prebuf   = (float*)alloc((size_t)BB*EE*4);
    int* sort_ind    = (int*)alloc((size_t)BB*4);
    int* dl          = (int*)alloc((size_t)BB*4);
    int* caps_sorted = (int*)alloc((size_t)BB*MAXLEN*4);

    k_sort<<<1, BB, 0, stream>>>(lens, caps, sort_ind, dl, caps_sorted,
                                 out + CAPS_OFF, out + SIND_OFF);

    auto cvt = [&](const float* s, __bf16* d, long n){
        k_f32_to_bf16<<<(int)((n + 255) / 256), 256, 0, stream>>>(s, d, n);
    };
    cvt(enc_att_w, wb_enc_att, (long)AA*ENCD);
    cvt(dec_att_w, wb_dec_att, (long)AA*HH);
    cvt(init_h_w,  wb_init_h,  (long)HH*ENCD);
    cvt(init_c_w,  wb_init_c,  (long)HH*ENCD);
    cvt(fbeta_w,   wb_fbeta,   (long)ENCD*HH);
    cvt(fc_w,      wb_fc,      (long)VV*HH);
    {
        long n = (long)(4*HH) * (EE + ENCD + HH);
        k_concat2_bf16<<<(int)((n + 255) / 256), 256, 0, stream>>>(w_ih, EE + ENCD, w_hh, HH, wb_lstm, n);
        n = (long)EE * (HH + ENCD);
        k_concat2_bf16<<<(int)((n + 255) / 256), 256, 0, stream>>>(lh_w, HH, lz_w, ENCD, wb_out, n);
    }
    k_addvec<<<(4*HH + 255) / 256, 256, 0, stream>>>(b_ih, b_hh, b_lstm, 4*HH);
    k_addvec<<<(EE + 255) / 256, 256, 0, stream>>>(lh_b, lz_b, b_out, EE);

    k_embgather<<<(int)(((long)BB*TT*EE + 255) / 256), 256, 0, stream>>>(emb, caps_sorted, emb_seq);
    k_mean<<<BB, 256, 0, stream>>>(enc, sort_ind, mean_enc);

    k_gemm_bf16<<<dim3(HH/64, 1), 256, 0, stream>>>(mean_enc, ENCD, nullptr, 1,
        wb_init_h, init_h_b, hbuf, HH, nullptr, 0, nullptr, 0, BB, HH, ENCD);
    k_gemm_bf16<<<dim3(HH/64, 1), 256, 0, stream>>>(mean_enc, ENCD, nullptr, 1,
        wb_init_c, init_c_b, cbuf, HH, nullptr, 0, nullptr, 0, BB, HH, ENCD);

    // att1 = enc_sorted @ enc_att_w^T + b   [25088, 512]
    k_gemm_bf16<<<dim3(AA/64, (BB*PP)/128), 256, 0, stream>>>(enc, ENCD, sort_ind, PP,
        wb_enc_att, enc_att_b, att1, AA, nullptr, 0, nullptr, 0, BB*PP, AA, ENCD);

    for (int t = 0; t < TT; t++) {
        k_gemm_bf16<<<dim3(AA/64, 1), 256, 0, stream>>>(hbuf, HH, nullptr, 1,
            wb_dec_att, dec_att_b, att2, AA, nullptr, 0, nullptr, 0, BB, AA, HH);

        k_score<<<(BB*PP*32 + 255) / 256, 256, 0, stream>>>(att1, att2, faw, fab, score);
        k_softmax<<<BB, 256, 0, stream>>>(score, alpha, out + ALPHA_OFF, dl, t);
        k_zctx<<<dim3(BB, ENCD/256), 256, 0, stream>>>(enc, sort_ind, alpha, zraw);

        k_gemm_bf16<<<dim3(ENCD/64, 1), 256, 0, stream>>>(hbuf, HH, nullptr, 1,
            wb_fbeta, fbeta_b, betapre, ENCD, nullptr, 0, nullptr, 0, BB, ENCD, HH);

        k_fuse1<<<(int)(((long)BB*(EE+ENCD+HH) + 255) / 256), 256, 0, stream>>>(
            emb_seq, t, betapre, zraw, hbuf, xcat, precat);

        k_gemm_bf16<<<dim3((4*HH)/64, 1), 256, 0, stream>>>(xcat, EE+ENCD+HH, nullptr, 1,
            wb_lstm, b_lstm, gates, 4*HH, nullptr, 0, nullptr, 0, BB, 4*HH, EE+ENCD+HH);

        k_cell<<<(BB*HH + 255) / 256, 256, 0, stream>>>(gates, hbuf, cbuf, precat, dl, t);

        k_gemm_bf16<<<dim3(EE/64, 1), 256, 0, stream>>>(precat, HH+ENCD, nullptr, 1,
            wb_out, b_out, prebuf, EE, emb_seq + (long)t*EE, (long)TT*EE, nullptr, 0,
            BB, EE, HH+ENCD);

        k_gemm_bf16<<<dim3((VV + 63)/64, 1), 256, 0, stream>>>(prebuf, EE, nullptr, 1,
            wb_fc, fc_b, out + PRED_OFF + (long)t*VV, (long)TT*VV, nullptr, 0, dl, t,
            BB, VV, EE);
    }
}
